// PestoModel_60619168416539
// MI455X (gfx1250) — matrix operand order repaired and checked by reading, not compiled, not run on hardware
//
#include <hip/hip_runtime.h>
#include <hip/hip_bf16.h>
#include <math.h>

// ---------------------------------------------------------------------------
// PestoModel for MI455X (gfx1250): edge-MLP GEMMs on v_wmma_f32_16x16x32_f16
// (f16 inputs, fp32 accumulate), scalar kernels for topk/attention/pooling.
// GEMM: templated N-tile count (pinned accumulators), double-buffered LDS A
// staging with vectorized fast path (global_load_b128 + ds_store_b64).
// ---------------------------------------------------------------------------

typedef __attribute__((ext_vector_type(16))) _Float16 v16h;
typedef __attribute__((ext_vector_type(4)))  _Float16 v4h;
typedef __attribute__((ext_vector_type(8)))  float    v8f;

#define N_NODES 4096
#define KNN     64
#define N1      4097              // padded (zero row 0)
#define E_EDGES (N1 * KNN)        // 262208
#define XE_LD   224               // padded K for edge features (193 used)
#define XE_K    193

// ============================ WMMA GEMM ====================================
// C[M x Nc] = act( A[M x Kdim] (lda) @ W[Kdim x Nc] + bias ), act: 0=none 1=ELU
// Weights in LDS as f16 transposed [n][k] (stride 232 halves: conflict-free,
// 16B aligned -> ds_load_b128 x2 per B fragment). A tiles staged per 32-wide
// K chunk into double-buffered LDS (stride 40 halves). Interior blocks take a
// branch-free vectorized staging path; masking exists only on the tail path.
#define MAX_KPAD 224
#define SW_LD    232              // sW row stride (halves)
#define SA_LD    40               // sA row stride (halves)

template <int NT>                 // number of 16-wide N tiles (Npad = NT*16)
__global__ __launch_bounds__(256) void gemm_wmma_kernel(
    const float* __restrict__ A, int lda, int M, int Kdim,
    const float* __restrict__ W, const float* __restrict__ bias,
    float* __restrict__ C, int ldc, int Nc, int act)
{
    constexpr int NPAD = NT * 16;
    __shared__ _Float16 sW[NPAD * SW_LD];     // [n][k]
    __shared__ _Float16 sA[2][128 * SA_LD];   // double-buffered [row][k-in-chunk]

    const int Kpad    = (Kdim + 31) & ~31;
    const int nChunks = Kpad >> 5;

    // ---- W -> LDS (f32 -> f16), zero-padded; no div/mod ----
#pragma unroll
    for (int n = 0; n < NPAD; ++n)
        for (int k = threadIdx.x; k < Kpad; k += 256) {
            float w = (k < Kdim && n < Nc) ? W[k * Nc + n] : 0.0f;
            sW[n * SW_LD + k] = (_Float16)w;
        }

    const int wave = threadIdx.x >> 5;   // 8 waves (wave32)
    const int lane = threadIdx.x & 31;
    const int l16  = lane & 15;
    const int hi   = lane >> 4;          // 0/1
    const int blockRow0 = blockIdx.x * 128;
    const int rowBase   = blockRow0 + wave * 16;
    const bool fullRows = (blockRow0 + 128) <= M;
    const bool ldaVec   = (lda & 3) == 0;

    // Stage one 128x32 A chunk into LDS buffer (f32 -> f16).
    auto stage = [&](int k0, _Float16* buf) {
        if (fullRows && ldaVec && (k0 + 32) <= Kdim) {
            // fast path: 4x float4 per thread, branch-free
#pragma unroll
            for (int i = 0; i < 4; ++i) {
                int u  = threadIdx.x + i * 256;   // quad index 0..1023
                int rr = u >> 3, qd = u & 7;
                const float4 f =
                    *(const float4*)(A + (size_t)(blockRow0 + rr) * lda + k0 + qd * 4);
                v4h h;
                h[0] = (_Float16)f.x; h[1] = (_Float16)f.y;
                h[2] = (_Float16)f.z; h[3] = (_Float16)f.w;
                *(v4h*)&buf[rr * SA_LD + qd * 4] = h;
            }
        } else {
            // tail path: masked scalar
            for (int idx = threadIdx.x; idx < 128 * 32; idx += 256) {
                int rr = idx >> 5, kk = idx & 31;
                int row = blockRow0 + rr;
                int k   = k0 + kk;
                float f = (row < M && k < Kdim) ? A[(size_t)row * lda + k] : 0.0f;
                buf[rr * SA_LD + kk] = (_Float16)f;
            }
        }
    };

    v8f acc[NT];
    v8f vzero = {};
#pragma unroll
    for (int t = 0; t < NT; ++t) acc[t] = vzero;

    stage(0, sA[0]);
    __syncthreads();                       // covers sW + first A chunk

    for (int c = 0; c < nChunks; ++c) {
        if (c + 1 < nChunks) stage(c + 1, sA[(c + 1) & 1]);   // prefetch next

        const int k0 = c << 5;
        const _Float16* ap = &sA[c & 1][(wave * 16 + l16) * SA_LD];
        v16h a;                            // a[0..15]=K 16*hi..16*hi+15
#pragma unroll
        for (int v = 0; v < 16; ++v) a[v] = ap[hi * 16 + v];

#pragma unroll
        for (int t = 0; t < NT; ++t) {
            const _Float16* bp = &sW[(t * 16 + l16) * SW_LD + k0 + hi * 16];
            v16h b;
#pragma unroll
            for (int v = 0; v < 16; ++v) b[v] = bp[v];
            acc[t] = __builtin_amdgcn_wmma_f32_16x16x32_f16(
                false, a, false, b, (short)0, acc[t], false, false);
        }
        __syncthreads();                   // next chunk staged & this one consumed
    }

    // ---- Epilogue: bias + ELU + store (C/D layout: M = v + 8*hi, N = l16) ---
#pragma unroll
    for (int t = 0; t < NT; ++t) {
        int col = t * 16 + l16;
#pragma unroll
        for (int v = 0; v < 8; ++v) {
            int row = rowBase + v + 8 * hi;
            if (row < M && col < Nc) {
                float r = acc[t][v];
                if (bias) r += bias[col];
                if (act)  r = (r > 0.0f) ? r : expm1f(r);
                C[(size_t)row * ldc + col] = r;
            }
        }
    }
}

// ============================ init / zero ==================================
__global__ void init_zero_kernel(float* q, float* p, int* ids, float* D,
                                 float* R, unsigned* dmax)
{
    int t = blockIdx.x * blockDim.x + threadIdx.x;
    if (t < N1 * 96) p[t] = 0.0f;           // p = zeros each call
    if (t < 32)      q[t] = 0.0f;           // q row 0
    if (t < KNN)   { ids[t] = 0; D[t] = 0.0f; }
    if (t < KNN * 3) R[t] = 0.0f;
    if (t == 0)     *dmax = 0u;
}

// ============================ top-K neighbors ==============================
__global__ __launch_bounds__(256) void topk_kernel(
    const float* __restrict__ pos, int* __restrict__ ids,
    float* __restrict__ Dout, float* __restrict__ Rout,
    unsigned* __restrict__ dmax_bits)
{
    __shared__ float sd[N_NODES];
    __shared__ float rv[256];
    __shared__ int   ri[256];

    const int i = blockIdx.x;
    const float xi = pos[i * 3], yi = pos[i * 3 + 1], zi = pos[i * 3 + 2];
    const float sqi = xi * xi + yi * yi + zi * zi;

    for (int j = threadIdx.x; j < N_NODES; j += 256) {
        float xj = pos[j * 3], yj = pos[j * 3 + 1], zj = pos[j * 3 + 2];
        float sqj = xj * xj + yj * yj + zj * zj;
        float dot = xi * xj + yi * yj + zi * zj;
        sd[j] = sqi + sqj - 2.0f * dot;
    }
    __syncthreads();

    float localMaxD = 0.0f;
    for (int it = 0; it < KNN; ++it) {
        float best = __builtin_inff(); int bi = N_NODES;
        for (int j = threadIdx.x; j < N_NODES; j += 256) {
            float v = sd[j];
            if (v < best || (v == best && j < bi)) { best = v; bi = j; }
        }
        rv[threadIdx.x] = best; ri[threadIdx.x] = bi;
        __syncthreads();
        for (int s = 128; s > 0; s >>= 1) {
            if (threadIdx.x < s) {
                float v2 = rv[threadIdx.x + s]; int i2 = ri[threadIdx.x + s];
                if (v2 < rv[threadIdx.x] ||
                    (v2 == rv[threadIdx.x] && i2 < ri[threadIdx.x])) {
                    rv[threadIdx.x] = v2; ri[threadIdx.x] = i2;
                }
            }
            __syncthreads();
        }
        if (threadIdx.x == 0) {
            int sel = ri[0];
            ids[(i + 1) * KNN + it] = sel;
            int idxv = sel - 1; if (idxv < 0) idxv += N_NODES;  // (sel-1) mod N
            float rx = pos[idxv * 3]     - xi;
            float ry = pos[idxv * 3 + 1] - yi;
            float rz = pos[idxv * 3 + 2] - zi;
            float d  = sqrtf(rx * rx + ry * ry + rz * rz);
            size_t e = (size_t)(i + 1) * KNN + it;
            Dout[e] = d;
            Rout[e * 3] = rx; Rout[e * 3 + 1] = ry; Rout[e * 3 + 2] = rz;
            localMaxD = fmaxf(localMaxD, d);
            sd[sel] = __builtin_inff();
        }
        __syncthreads();
    }
    if (threadIdx.x == 0) atomicMax(dmax_bits, __float_as_uint(localMaxD));
}

__global__ void finalize_rd_kernel(float* D, float* R,
                                   const unsigned* dmax_bits)
{
    int e = blockIdx.x * blockDim.x + threadIdx.x;
    if (e >= E_EDGES || e < KNN) return;    // keep row 0 zero
    float dmax = __uint_as_float(*dmax_bits);
    float d = D[e];
    float dadj = d + ((d < 0.01f) ? dmax : 0.0f);
    D[e] = dadj;
    float inv = 1.0f / dadj;
    R[e * 3] *= inv; R[e * 3 + 1] *= inv; R[e * 3 + 2] *= inv;
}

// ======================== node features / z ================================
// out[n] = [ q[n+off] (32), ||p[n+off]|| over xyz (32) ]
__global__ void build_xn_kernel(const float* __restrict__ q,
                                const float* __restrict__ p,
                                float* __restrict__ out, int rows, int off)
{
    int t = blockIdx.x * blockDim.x + threadIdx.x;
    if (t >= rows * 64) return;
    int n = t >> 6, c = t & 63, i = n + off;
    float v;
    if (c < 32) v = q[i * 32 + c];
    else {
        int s = c - 32;
        float a = p[(i * 3 + 0) * 32 + s];
        float b = p[(i * 3 + 1) * 32 + s];
        float d = p[(i * 3 + 2) * 32 + s];
        v = sqrtf(a * a + b * b + d * d);
    }
    out[t] = v;
}

// ============================ edge features ================================
// X_e row (stride 224): [d | X_n(64) | q_nn(32) | |p_nn|(32) | pr(32) | pnr(32) | 0-pad]
__global__ void build_xe_kernel(const float* __restrict__ Xn,
                                const float* __restrict__ q,
                                const float* __restrict__ p,
                                const int*   __restrict__ ids,
                                const float* __restrict__ D,
                                const float* __restrict__ R,
                                float* __restrict__ Xe, int total)
{
    int t = blockIdx.x * blockDim.x + threadIdx.x;
    if (t >= total) return;
    int e = t / XE_LD, col = t % XE_LD;
    float v = 0.0f;
    if (col == 0) v = D[e];
    else if (col < 65) v = Xn[(e >> 6) * 64 + (col - 1)];
    else if (col < 193) {
        int i = e >> 6;
        int nid = ids[e];
        int s   = (col - 65) & 31;
        int grp = (col - 65) >> 5;       // 0:q_nn 1:|p_nn| 2:pr 3:pnr
        if (grp == 0) v = q[nid * 32 + s];
        else if (grp == 1) {
            float a = p[(nid * 3 + 0) * 32 + s];
            float b = p[(nid * 3 + 1) * 32 + s];
            float c2 = p[(nid * 3 + 2) * 32 + s];
            v = sqrtf(a * a + b * b + c2 * c2);
        } else {
            float rx = R[(size_t)e * 3], ry = R[(size_t)e * 3 + 1],
                  rz = R[(size_t)e * 3 + 2];
            int base = ((grp == 2) ? i : nid) * 3;
            v = p[(base + 0) * 32 + s] * rx + p[(base + 1) * 32 + s] * ry +
                p[(base + 2) * 32 + s] * rz;
        }
    }
    Xe[t] = v;
}

// ============================ attention ====================================
// One block per node: softmax over 64 (q-heads) / 192 (p-heads), then Zq/Zp.
__global__ __launch_bounds__(64) void attention_kernel(
    const float* __restrict__ Qm,    // N1 x 16 (12 used: [a(2)][h(2)][k(3)])
    const float* __restrict__ Kq,    // E x 16 (3 used)
    const float* __restrict__ Kp9,   // E x 16 (9 used)
    const float* __restrict__ V,     // E x 64 (V0|V1)
    const float* __restrict__ R,     // E x 3
    const float* __restrict__ p,     // N1 x 3 x 32
    const int*   __restrict__ ids,   // E
    float* __restrict__ Zq,          // N1 x 64
    float* __restrict__ Zp)          // (N1*3) x 64
{
    __shared__ float sQ[12];
    __shared__ float sMq[2][64];
    __shared__ float sMp[2][192];

    const int i = blockIdx.x;
    const int j = threadIdx.x;                 // 0..63
    const size_t e0 = (size_t)i * KNN;
    const float isdk = 0.5773502691896258f;    // 1/sqrt(NK=3)

    if (j < 12) sQ[j] = Qm[i * 16 + j];
    __syncthreads();

    {
        float k0 = Kq[(e0 + j) * 16 + 0];
        float k1 = Kq[(e0 + j) * 16 + 1];
        float k2 = Kq[(e0 + j) * 16 + 2];
        for (int h = 0; h < 2; ++h)
            sMq[h][j] = (sQ[h * 3] * k0 + sQ[h * 3 + 1] * k1 + sQ[h * 3 + 2] * k2) * isdk;
    }
    for (int g = 0; g < 3; ++g) {
        float k0 = Kp9[(e0 + j) * 16 + g * 3 + 0];
        float k1 = Kp9[(e0 + j) * 16 + g * 3 + 1];
        float k2 = Kp9[(e0 + j) * 16 + g * 3 + 2];
        for (int h = 0; h < 2; ++h)
            sMp[h][g * 64 + j] =
                (sQ[6 + h * 3] * k0 + sQ[6 + h * 3 + 1] * k1 + sQ[6 + h * 3 + 2] * k2) * isdk;
    }
    __syncthreads();

    if (j < 2) {                                // softmax rows of Mq
        float m = -__builtin_inff();
        for (int t = 0; t < 64; ++t) m = fmaxf(m, sMq[j][t]);
        float s = 0.0f;
        for (int t = 0; t < 64; ++t) { float w = expf(sMq[j][t] - m); sMq[j][t] = w; s += w; }
        float inv = 1.0f / s;
        for (int t = 0; t < 64; ++t) sMq[j][t] *= inv;
    } else if (j < 4) {                         // softmax rows of Mp
        int h = j - 2;
        float m = -__builtin_inff();
        for (int t = 0; t < 192; ++t) m = fmaxf(m, sMp[h][t]);
        float s = 0.0f;
        for (int t = 0; t < 192; ++t) { float w = expf(sMp[h][t] - m); sMp[h][t] = w; s += w; }
        float inv = 1.0f / s;
        for (int t = 0; t < 192; ++t) sMp[h][t] *= inv;
    }
    __syncthreads();

    const int h  = j >> 5;
    const int s_ = j & 31;
    {
        float acc = 0.0f;
        for (int t = 0; t < 64; ++t) acc += sMq[h][t] * V[(e0 + t) * 64 + s_];
        Zq[(size_t)i * 64 + j] = acc;
    }
    {
        float a0 = 0.0f, a1 = 0.0f, a2 = 0.0f;
        float pi0 = p[(i * 3 + 0) * 32 + s_];
        float pi1 = p[(i * 3 + 1) * 32 + s_];
        float pi2 = p[(i * 3 + 2) * 32 + s_];
        for (int t = 0; t < 64; ++t) {
            size_t e = e0 + t;
            float w0 = sMp[h][t];         // V1 * r term
            float w1 = sMp[h][64 + t];    // p broadcast term
            float w2 = sMp[h][128 + t];   // p_nn term
            float v1 = V[e * 64 + 32 + s_];
            float rx = R[e * 3], ry = R[e * 3 + 1], rz = R[e * 3 + 2];
            int nid = ids[e];
            a0 += w0 * (v1 * rx) + w1 * pi0 + w2 * p[(nid * 3 + 0) * 32 + s_];
            a1 += w0 * (v1 * ry) + w1 * pi1 + w2 * p[(nid * 3 + 1) * 32 + s_];
            a2 += w0 * (v1 * rz) + w1 * pi2 + w2 * p[(nid * 3 + 2) * 32 + s_];
        }
        Zp[((size_t)i * 3 + 0) * 64 + j] = a0;
        Zp[((size_t)i * 3 + 1) * 64 + j] = a1;
        Zp[((size_t)i * 3 + 2) * 64 + j] = a2;
    }
}

// ============================ state update =================================
__global__ void update_qp_kernel(float* q, float* p,
                                 const float* dq, const float* dp)
{
    int t = blockIdx.x * blockDim.x + threadIdx.x;
    const int nq = N1 * 32, np = N1 * 96;
    if (t < nq) {
        int i = t >> 5;
        q[t] = (i == 0) ? 0.0f : (q[t] + dq[t]);
    } else if (t < nq + np) {
        int u = t - nq;
        int i = u / 96;
        p[u] = (i == 0) ? 0.0f : (p[u] + dp[u]);
    }
}

// ============================ residue pooling ==============================
// Group softmax per residue (non-matching F ~= -1e6 => exp underflows to 0).
__global__ __launch_bounds__(128) void pool_kernel(
    const float* __restrict__ s8,       // 4096 x 16 (8 used)
    const float* __restrict__ q,        // N1 x 32 (qa = q[n+1])
    const float* __restrict__ p,        // N1 x 3 x 32
    const int*   __restrict__ res_map,  // 4096
    float* __restrict__ qh,             // 512 x 128
    float* __restrict__ ph)             // (512*3) x 128
{
    __shared__ int   cnt;
    __shared__ int   nl[64];
    __shared__ float w[64][8];

    const int r = blockIdx.x;
    if (threadIdx.x == 0) cnt = 0;
    __syncthreads();
    for (int n = threadIdx.x; n < N_NODES; n += 128) {
        if (res_map[n] - 1 == r) {
            int k = atomicAdd(&cnt, 1);
            if (k < 64) nl[k] = n;
        }
    }
    __syncthreads();
    const int G = (cnt < 64) ? cnt : 64;

    if (threadIdx.x < 8) {
        int c = threadIdx.x;
        float m = -__builtin_inff();
        for (int g = 0; g < G; ++g) m = fmaxf(m, s8[nl[g] * 16 + c]);
        float sum = 0.0f;
        for (int g = 0; g < G; ++g) {
            float e = expf(s8[nl[g] * 16 + c] - m);
            w[g][c] = e; sum += e;
        }
        float inv = (sum > 0.0f) ? (1.0f / sum) : 0.0f;
        for (int g = 0; g < G; ++g) w[g][c] *= inv;
    }
    __syncthreads();

    // qh[r][s*4 + h'] with channel a=0 => c = 2h'
    {
        int o = threadIdx.x;            // 0..127
        int s_ = o >> 2, hp = o & 3;
        float acc = 0.0f;
        for (int g = 0; g < G; ++g)
            acc += w[g][2 * hp] * q[(nl[g] + 1) * 32 + s_];
        qh[(size_t)r * 128 + o] = acc;
    }
    // ph[(r*3+c)][s*4 + h'] with channel a=1 => c = 2h'+1
    for (int o = threadIdx.x; o < 384; o += 128) {
        int c = o >> 7, rem = o & 127;
        int s_ = rem >> 2, hp = rem & 3;
        float acc = 0.0f;
        for (int g = 0; g < G; ++g)
            acc += w[g][2 * hp + 1] * p[((nl[g] + 1) * 3 + c) * 32 + s_];
        ph[((size_t)r * 3 + c) * 128 + rem] = acc;
    }
}

// ============================ final output =================================
__global__ void final_kernel(const float* __restrict__ qr,
                             const float* __restrict__ prv,
                             float* __restrict__ out)
{
    int t = blockIdx.x * blockDim.x + threadIdx.x;
    if (t >= 512 * 128) return;
    int r = t >> 7, c = t & 127;
    float v;
    if (c < 64) v = qr[r * 64 + c];
    else {
        int s = c - 64;
        float a = prv[(r * 3 + 0) * 64 + s];
        float b = prv[(r * 3 + 1) * 64 + s];
        float d = prv[(r * 3 + 2) * 64 + s];
        v = sqrtf(a * a + b * b + d * d);
    }
    out[t] = v;
}

// ============================ host side ====================================
extern "C" void kernel_launch(void* const* d_in, const int* in_sizes, int n_in,
                              void* d_out, int out_size, void* d_ws, size_t ws_size,
                              hipStream_t stream)
{
    (void)in_sizes; (void)out_size; (void)ws_size;

    // ---- input mapping (jax pytree flatten: dict keys sorted) ----
    // 0:x 1:pos | params: em(W,b x3)=2..7 | spl.sam=8..13 spl.zdm=14..19
    // spl.zdm_vec.W=20 | sum[l] base=21+31l: epkm(0..5) eqkm(6..11) evm(12..17)
    // nqm(18..23) ppm.W(24) qpm(25..30) | res_map = last
    auto P = [&](int i) { return (const float*)d_in[i]; };
    const float* x    = P(0);
    const float* pos  = P(1);
    const int* res_map = (const int*)d_in[n_in - 1];

    const float *emW[3]  = {P(2),  P(4),  P(6)},  *emB[3]  = {P(3),  P(5),  P(7)};
    const float *samW[3] = {P(8),  P(10), P(12)}, *samB[3] = {P(9),  P(11), P(13)};
    const float *zdmW[3] = {P(14), P(16), P(18)}, *zdmB[3] = {P(15), P(17), P(19)};
    const float *zdmVecW = P(20);

    // ---- workspace carve ----
    char* base = (char*)d_ws;
    size_t off = 0;
    auto alloc = [&](size_t bytes) {
        void* ptr = base + off;
        off = (off + bytes + 255) & ~(size_t)255;
        return ptr;
    };
    int*      ids  = (int*)     alloc((size_t)E_EDGES * sizeof(int));
    float*    D    = (float*)   alloc((size_t)E_EDGES * sizeof(float));
    float*    R    = (float*)   alloc((size_t)E_EDGES * 3 * sizeof(float));
    unsigned* dmax = (unsigned*)alloc(sizeof(unsigned));
    float*    q    = (float*)   alloc((size_t)N1 * 32 * sizeof(float));
    float*    p    = (float*)   alloc((size_t)N1 * 96 * sizeof(float));
    float*    Xn   = (float*)   alloc((size_t)N1 * 64 * sizeof(float));
    float*    Qm   = (float*)   alloc((size_t)N1 * 16 * sizeof(float));
    float*    Xe   = (float*)   alloc((size_t)E_EDGES * XE_LD * sizeof(float));
    float*    H1   = (float*)   alloc((size_t)E_EDGES * 64 * sizeof(float));
    float*    H2   = (float*)   alloc((size_t)E_EDGES * 64 * sizeof(float));
    float*    Kq   = (float*)   alloc((size_t)E_EDGES * 16 * sizeof(float));
    float*    Kp   = (float*)   alloc((size_t)E_EDGES * 16 * sizeof(float));
    float*    V    = (float*)   alloc((size_t)E_EDGES * 64 * sizeof(float));
    float*    Zq   = (float*)   alloc((size_t)N1 * 64 * sizeof(float));
    float*    Zp   = (float*)   alloc((size_t)N1 * 3 * 64 * sizeof(float));
    float*    dq   = (float*)   alloc((size_t)N1 * 32 * sizeof(float));
    float*    dp   = (float*)   alloc((size_t)N1 * 3 * 32 * sizeof(float));
    float*    z    = (float*)   alloc((size_t)N_NODES * 64 * sizeof(float));
    float*    s8   = (float*)   alloc((size_t)N_NODES * 16 * sizeof(float));
    float*    qh   = (float*)   alloc((size_t)512 * 128 * sizeof(float));
    float*    ph   = (float*)   alloc((size_t)512 * 3 * 128 * sizeof(float));
    float*    qr   = (float*)   alloc((size_t)512 * 64 * sizeof(float));
    float*    prv  = (float*)   alloc((size_t)512 * 3 * 64 * sizeof(float));

    auto gemm = [&](const float* A, int lda, int M, int Kdim, const float* W,
                    const float* b, float* C, int ldc, int Nc, int act) {
        int grid = (M + 127) / 128;
        int npad = (Nc + 15) & ~15;
        if (npad == 16)
            gemm_wmma_kernel<1><<<grid, 256, 0, stream>>>(A, lda, M, Kdim, W, b, C, ldc, Nc, act);
        else if (npad == 32)
            gemm_wmma_kernel<2><<<grid, 256, 0, stream>>>(A, lda, M, Kdim, W, b, C, ldc, Nc, act);
        else
            gemm_wmma_kernel<4><<<grid, 256, 0, stream>>>(A, lda, M, Kdim, W, b, C, ldc, Nc, act);
    };

    // ---- graph construction ----
    init_zero_kernel<<<(N1 * 96 + 255) / 256, 256, 0, stream>>>(q, p, ids, D, R, dmax);
    topk_kernel<<<N_NODES, 256, 0, stream>>>(pos, ids, D, R, dmax);
    finalize_rd_kernel<<<(E_EDGES + 255) / 256, 256, 0, stream>>>(D, R, dmax);

    // ---- embedding: q[1..N] = em(x), ELU between layers ----
    gemm(x,  30, N_NODES, 30, emW[0], emB[0], H1, 32, 32, 1);
    gemm(H1, 32, N_NODES, 32, emW[1], emB[1], H2, 32, 32, 1);
    gemm(H2, 32, N_NODES, 32, emW[2], emB[2], q + 32, 32, 32, 0);

    // ---- 4 message-passing layers ----
    for (int l = 0; l < 4; ++l) {
        const int b0 = 21 + 31 * l;
        const float *epkmW[3] = {P(b0+0),  P(b0+2),  P(b0+4)};
        const float *epkmB[3] = {P(b0+1),  P(b0+3),  P(b0+5)};
        const float *eqkmW[3] = {P(b0+6),  P(b0+8),  P(b0+10)};
        const float *eqkmB[3] = {P(b0+7),  P(b0+9),  P(b0+11)};
        const float *evmW[3]  = {P(b0+12), P(b0+14), P(b0+16)};
        const float *evmB[3]  = {P(b0+13), P(b0+15), P(b0+17)};
        const float *nqmW[3]  = {P(b0+18), P(b0+20), P(b0+22)};
        const float *nqmB[3]  = {P(b0+19), P(b0+21), P(b0+23)};
        const float *ppmW     =  P(b0+24);
        const float *qpmW[3]  = {P(b0+25), P(b0+27), P(b0+29)};
        const float *qpmB[3]  = {P(b0+26), P(b0+28), P(b0+30)};

        build_xn_kernel<<<(N1 * 64 + 255) / 256, 256, 0, stream>>>(q, p, Xn, N1, 0);

        gemm(Xn, 64, N1, 64, nqmW[0], nqmB[0], H1, 32, 32, 1);
        gemm(H1, 32, N1, 32, nqmW[1], nqmB[1], H2, 32, 32, 1);
        gemm(H2, 32, N1, 32, nqmW[2], nqmB[2], Qm, 16, 12, 0);

        int xe_total = E_EDGES * XE_LD;
        build_xe_kernel<<<(xe_total + 255) / 256, 256, 0, stream>>>(
            Xn, q, p, ids, D, R, Xe, xe_total);

        gemm(Xe, XE_LD, E_EDGES, XE_K, eqkmW[0], eqkmB[0], H1, 32, 32, 1);
        gemm(H1, 32,    E_EDGES, 32,   eqkmW[1], eqkmB[1], H2, 32, 32, 1);
        gemm(H2, 32,    E_EDGES, 32,   eqkmW[2], eqkmB[2], Kq, 16, 3,  0);

        gemm(Xe, XE_LD, E_EDGES, XE_K, epkmW[0], epkmB[0], H1, 32, 32, 1);
        gemm(H1, 32,    E_EDGES, 32,   epkmW[1], epkmB[1], H2, 32, 32, 1);
        gemm(H2, 32,    E_EDGES, 32,   epkmW[2], epkmB[2], Kp, 16, 9,  0);

        gemm(Xe, XE_LD, E_EDGES, XE_K, evmW[0], evmB[0], H1, 64, 64, 1);
        gemm(H1, 64,    E_EDGES, 64,   evmW[1], evmB[1], H2, 64, 64, 1);
        gemm(H2, 64,    E_EDGES, 64,   evmW[2], evmB[2], V,  64, 64, 0);

        attention_kernel<<<N1, 64, 0, stream>>>(Qm, Kq, Kp, V, R, p, ids, Zq, Zp);

        gemm(Zq, 64, N1,     64, qpmW[0], qpmB[0], H1, 32, 32, 1);
        gemm(H1, 32, N1,     32, qpmW[1], qpmB[1], H2, 32, 32, 1);
        gemm(H2, 32, N1,     32, qpmW[2], qpmB[2], dq, 32, 32, 0);
        gemm(Zp, 64, N1 * 3, 64, ppmW, nullptr,    dp, 32, 32, 0);

        update_qp_kernel<<<(N1 * 128 + 255) / 256, 256, 0, stream>>>(q, p, dq, dp);
    }

    // ---- residue pooling head ----
    build_xn_kernel<<<(N_NODES * 64 + 255) / 256, 256, 0, stream>>>(q, p, z, N_NODES, 1);

    gemm(z,  64, N_NODES, 64, samW[0], samB[0], H1, 32, 32, 1);
    gemm(H1, 32, N_NODES, 32, samW[1], samB[1], H2, 32, 32, 1);
    gemm(H2, 32, N_NODES, 32, samW[2], samB[2], s8, 16, 8,  0);

    pool_kernel<<<512, 128, 0, stream>>>(s8, q, p, res_map, qh, ph);

    gemm(qh, 128, 512,  128, zdmW[0], zdmB[0], H1, 32, 32, 1);
    gemm(H1, 32,  512,  32,  zdmW[1], zdmB[1], H2, 32, 32, 1);
    gemm(H2, 32,  512,  32,  zdmW[2], zdmB[2], qr, 64, 64, 0);
    gemm(ph, 128, 1536, 128, zdmVecW, nullptr, prv, 64, 64, 0);

    final_kernel<<<(512 * 128 + 255) / 256, 256, 0, stream>>>(qr, prv, (float*)d_out);
}
